// DecoderLayer_54125177864203
// MI455X (gfx1250) — matrix operand order repaired and checked by reading, not compiled, not run on hardware
//
#include <hip/hip_runtime.h>
#include <stdint.h>

// ---------------------------------------------------------------------------
// Types & WMMA helpers (CDNA5 / gfx1250, wave32)
// ---------------------------------------------------------------------------
typedef __bf16 bf16;
typedef __attribute__((ext_vector_type(16))) __bf16 v16bf;
typedef __attribute__((ext_vector_type(8)))  float  v8f;
typedef __attribute__((ext_vector_type(4)))  int    v4i;

union U16bf { v16bf v; uint32_t u[8]; uint4 q[2]; bf16 e[16]; };
union U8f   { v8f v; float f[8]; };

__device__ __forceinline__ v8f wmma_bf16f32(v16bf a, v16bf b, v8f c) {
  // D = A(16x32 bf16) * B(32x16 bf16) + C(16x16 f32)
  return __builtin_amdgcn_wmma_f32_16x16x32_bf16(
      /*neg_a=*/false, a, /*neg_b=*/false, b,
      /*c_mod=*/(short)0, c, /*reuse_a=*/false, /*reuse_b=*/false);
}

// A-matrix 16x32 bf16 fragment (ISA 7.12.2):
// lanes 0-15: row M=lane, k = {0..7,16..23}; lanes 16-31: same rows, k = {8..15,24..31}.
// k-pairs (2j,2j+1) are adjacent -> one 32-bit load per VGPR from a row-major row.
__device__ __forceinline__ v16bf load_a_frag(const bf16* rowp, int lh) {
  U16bf r;
  const uint32_t* p = (const uint32_t*)rowp;
#pragma unroll
  for (int j = 0; j < 4; ++j) r.u[j] = p[j + 4 * lh];         // k = 2j+8*lh
#pragma unroll
  for (int j = 0; j < 4; ++j) r.u[4 + j] = p[8 + j + 4 * lh]; // k = 16+2j+8*lh
  return r.v;
}

// B-matrix 32x16 bf16 fragment: lane holds column (lane&15), k = {8h..8h+7,
// 16+8h..16+8h+7} with h=lane>>4. colp points at k=0 of that column (k contiguous).
__device__ __forceinline__ v16bf load_b_frag(const bf16* colp, int lh) {
  U16bf r;
  const uint4* p = (const uint4*)(colp + 8 * lh);
  r.q[0] = p[0];
  r.q[1] = p[2];
  return r.v;
}

// ---------------------------------------------------------------------------
// Async global -> LDS copy (16 bytes / lane). Uses the CDNA5 ASYNCcnt-tracked
// GLOBAL_LOAD_ASYNC_TO_LDS_B128 path when the builtin exists; otherwise a
// synchronous copy (barrier scheme below is correct for both).
// ---------------------------------------------------------------------------
#if defined(__has_builtin)
#if __has_builtin(__builtin_amdgcn_global_load_async_to_lds_b128)
#define HAVE_ASYNC_LDS 1
#endif
#endif
#ifndef HAVE_ASYNC_LDS
#define HAVE_ASYNC_LDS 0
#endif

__device__ __forceinline__ void async_copy16(const bf16* gsrc, bf16* ldst) {
#if HAVE_ASYNC_LDS
  // Param pointee type is a 128-bit int vector; pass generic pointers and let
  // clang insert the global/LDS addrspacecasts.
  __builtin_amdgcn_global_load_async_to_lds_b128((v4i*)gsrc, (v4i*)ldst,
                                                 /*offset=*/0, /*cpol=*/0);
#else
  *(uint4*)ldst = *(const uint4*)gsrc;
#endif
}

__device__ __forceinline__ void wait_async_all() {
#if defined(__has_builtin)
#if __has_builtin(__builtin_amdgcn_s_wait_asynccnt)
  __builtin_amdgcn_s_wait_asynccnt(0);
  __builtin_amdgcn_fence(__ATOMIC_ACQUIRE, "workgroup");
  return;
#endif
#endif
  asm volatile("s_wait_asynccnt 0" ::: "memory");
}

// ---------------------------------------------------------------------------
// Conversion kernels (fp32 -> bf16), weights stored TRANSPOSED: Bt[n][k]
// ---------------------------------------------------------------------------
__global__ __launch_bounds__(256) void cvt_f32_bf16_kernel(
    const float* __restrict__ in, bf16* __restrict__ out, int n) {
  int i = blockIdx.x * 256 + threadIdx.x;
  if (i < n) out[i] = (bf16)in[i];
}

// W[H=16, E=1024, D=64] fp32 -> Wt[H*D, E] bf16 (n=(h,d) rows, k=e cols)
__global__ __launch_bounds__(256) void cvt_wqkv_t_kernel(
    const float* __restrict__ in, bf16* __restrict__ out) {
  int i = blockIdx.x * 256 + threadIdx.x;  // < 16*1024*64
  int h = i >> 16, e = (i >> 6) & 1023, d = i & 63;
  out[(((size_t)(h << 6) + d) << 10) + e] = (bf16)in[i];
}

// in [R][C] fp32 -> out [C][R] bf16
__global__ __launch_bounds__(256) void cvt_mat_t_kernel(
    const float* __restrict__ in, bf16* __restrict__ out, int R, int C) {
  int i = blockIdx.x * 256 + threadIdx.x;
  int r = i / C, c = i - r * C;
  out[(size_t)c * R + r] = (bf16)in[i];
}

// ---------------------------------------------------------------------------
// LayerNorm: fp32 [rows, E] -> bf16 [rows, E]
// ---------------------------------------------------------------------------
__global__ __launch_bounds__(256) void ln_bf16_kernel(
    const float* __restrict__ x, const float* __restrict__ g,
    const float* __restrict__ b, bf16* __restrict__ out, int E) {
  const int row = blockIdx.x, tid = threadIdx.x;
  const float* xr = x + (size_t)row * E;
  float s = 0.f, ss = 0.f;
  for (int i = tid; i < E; i += 256) { float v = xr[i]; s += v; ss += v * v; }
#pragma unroll
  for (int off = 16; off >= 1; off >>= 1) {
    s += __shfl_xor(s, off, 32);
    ss += __shfl_xor(ss, off, 32);
  }
  __shared__ float sh[16];
  int wid = tid >> 5, lane = tid & 31;
  if (lane == 0) { sh[wid] = s; sh[8 + wid] = ss; }
  __syncthreads();
  float ts = 0.f, tss = 0.f;
#pragma unroll
  for (int w = 0; w < 8; ++w) { ts += sh[w]; tss += sh[8 + w]; }
  float mu = ts / (float)E;
  float var = tss / (float)E - mu * mu;
  float inv = rsqrtf(var + 1e-5f);
  for (int i = tid; i < E; i += 256)
    out[(size_t)row * E + i] = (bf16)((xr[i] - mu) * inv * g[i] + b[i]);
}

// ---------------------------------------------------------------------------
// bf16 GEMM: C[M,N] = A[M,K] * Bt[N,K]^T + bias, double-buffered async tiles.
// Block: 256 threads = 8 waves (4x2), block tile 128x128, BK=32.
// Wave tile 32x64: 2 m-frags x 4 n-frags -> 8 WMMA per k-stage.
// ---------------------------------------------------------------------------
#define LDT 40  // padded k-stride (elems); 80B per row, multiple of 16B

enum { MODE_QKV = 0, MODE_RES = 1, MODE_GELU = 2, MODE_VT = 3 };

template <int MODE>
__global__ __launch_bounds__(256) void gemm_bf16_kernel(
    const bf16* __restrict__ A, const bf16* __restrict__ Bt,
    const float* __restrict__ bias, const float* __restrict__ resid,
    void* __restrict__ outp, int M, int N, int K) {
  __shared__ __align__(16) bf16 As[2][128 * LDT];
  __shared__ __align__(16) bf16 Bs[2][128 * LDT];  // Bs[n][k]

  const int tid = threadIdx.x;
  const int lane = tid & 31, wid = tid >> 5;
  const int lh = lane >> 4, l15 = lane & 15;
  const int wrow = wid >> 1, wcol = wid & 1;
  const int bm = blockIdx.y * 128, bn = blockIdx.x * 128;

  U8f acc[2][4];
#pragma unroll
  for (int i = 0; i < 2; ++i)
#pragma unroll
    for (int j = 0; j < 4; ++j)
#pragma unroll
      for (int k = 0; k < 8; ++k) acc[i][j].f[k] = 0.f;

  const int row = tid >> 1, seg = (tid & 1) * 16;  // per-thread copy slot
  auto issue_stage = [&](int k0, int buf) {
    const bf16* ga = A + (size_t)(bm + row) * K + k0 + seg;
    const bf16* gb = Bt + (size_t)(bn + row) * K + k0 + seg;
    bf16* la = &As[buf][row * LDT + seg];
    bf16* lb = &Bs[buf][row * LDT + seg];
    async_copy16(ga, la);
    async_copy16(ga + 8, la + 8);
    async_copy16(gb, lb);
    async_copy16(gb + 8, lb + 8);
  };

  const int nst = K >> 5;
  issue_stage(0, 0);
  for (int si = 0; si < nst; ++si) {
    const int cur = si & 1;
    wait_async_all();
    __syncthreads();
    if (si + 1 < nst) issue_stage((si + 1) << 5, cur ^ 1);

    v16bf af0 = load_a_frag(&As[cur][(wrow * 32 + l15) * LDT], lh);
    v16bf af1 = load_a_frag(&As[cur][(wrow * 32 + 16 + l15) * LDT], lh);
#pragma unroll
    for (int ni = 0; ni < 4; ++ni) {
      v16bf bfrag = load_b_frag(&Bs[cur][(wcol * 64 + ni * 16 + l15) * LDT], lh);
      acc[0][ni].v = wmma_bf16f32(af0, bfrag, acc[0][ni].v);
      acc[1][ni].v = wmma_bf16f32(af1, bfrag, acc[1][ni].v);
    }
  }

  // Epilogue. C layout: VGPR r, lanes 0-15 -> row r, lanes 16-31 -> row r+8;
  // col = lane&15.
#pragma unroll
  for (int mi = 0; mi < 2; ++mi)
#pragma unroll
    for (int ni = 0; ni < 4; ++ni)
#pragma unroll
      for (int r = 0; r < 8; ++r) {
        int m = bm + wrow * 32 + mi * 16 + r + 8 * lh;
        int n = bn + wcol * 64 + ni * 16 + l15;
        float v = acc[mi][ni].f[r] + bias[n];
        if (MODE == MODE_QKV) {
          // m=(b,s), n=(h,d) -> bf16 out [B,H,S,D]
          int bb = m >> 10, s = m & 1023, hh = n >> 6, d = n & 63;
          ((bf16*)outp)[((((size_t)bb * 16 + hh) << 10) + s) * 64 + d] = (bf16)v;
        } else if (MODE == MODE_VT) {
          // m=(b,t), n=(h,d) -> bf16 out [B,H,D,T] (pre-transposed V)
          int bb = m >> 10, t = m & 1023, hh = n >> 6, d = n & 63;
          ((bf16*)outp)[((((size_t)bb * 16 + hh) * 64 + d) << 10) + t] = (bf16)v;
        } else if (MODE == MODE_RES) {
          size_t off = (size_t)m * N + n;
          ((float*)outp)[off] = v + resid[off];
        } else {  // MODE_GELU (exact, erf-based)
          float gv = 0.5f * v * (1.f + erff(v * 0.70710678118654752f));
          ((bf16*)outp)[(size_t)m * N + n] = (bf16)gv;
        }
      }
}

// ---------------------------------------------------------------------------
// Flash attention (no 1/sqrt(d) scale, faithful to reference).
// Q,K: bf16 [B,H,S|T,D=64]; V: bf16 [B,H,D,T] (pre-transposed); Out: [B,S,H*D].
// Block = 128 threads (4 waves); block owns 64 query rows, wave owns 16.
// Key/value chunks of 64 double-buffered through LDS via async copies.
// ---------------------------------------------------------------------------
#define KSTR 72  // padded stride (elems): 144B rows, multiple of 16B

__global__ __launch_bounds__(128) void attn_kernel(
    const bf16* __restrict__ Q, const bf16* __restrict__ Km,
    const bf16* __restrict__ Vtg, bf16* __restrict__ Out,
    int Slen, int Tlen, int causal) {
  const int Hh = 16, D = 64;
  const int tid = threadIdx.x, wid = tid >> 5, lane = tid & 31;
  const int lh = lane >> 4, l15 = lane & 15;
  const int h = blockIdx.y, b = blockIdx.z;
  const int qbase = blockIdx.x * 64;
  const int qrow0 = qbase + wid * 16;

  const size_t bh = (size_t)b * Hh + h;
  const bf16* Qp = Q + bh * (size_t)Slen * D;
  const bf16* Kp = Km + bh * (size_t)Tlen * D;   // [t][d]
  const bf16* Vp = Vtg + bh * (size_t)D * Tlen;  // [d][t]

  __shared__ __align__(16) bf16 Ks[2][64 * KSTR];   // Ks[t][d]
  __shared__ __align__(16) bf16 Vt[2][64 * KSTR];   // Vt[d][t]
  __shared__ __align__(16) bf16 Ps[4 * 16 * KSTR];  // per-wave P tile [16][64]
  bf16* Pw = &Ps[wid * 16 * KSTR];

  // Q fragments (A-layout), loaded once: row = qrow0 + (lane&15)
  v16bf aq0, aq1;
  {
    const bf16* qr = Qp + (size_t)(qrow0 + l15) * D;
    aq0 = load_a_frag(qr, lh);       // d in [0,32)
    aq1 = load_a_frag(qr + 32, lh);  // d in [32,64)
  }

  U8f o[4];
#pragma unroll
  for (int j = 0; j < 4; ++j)
#pragma unroll
    for (int r = 0; r < 8; ++r) o[j].f[r] = 0.f;
  float mrow[8], lrow[8];
#pragma unroll
  for (int r = 0; r < 8; ++r) { mrow[r] = -3.0e38f; lrow[r] = 0.f; }

  const int crow = tid >> 1, cseg = (tid & 1) * 32;  // per-thread copy slot
  auto issue_chunk = [&](int tc, int buf) {
    const bf16* ks = Kp + (size_t)(tc + crow) * D + cseg;   // K row t=crow
    const bf16* vs = Vp + (size_t)crow * Tlen + tc + cseg;  // V^T row d=crow
    bf16* kd = &Ks[buf][crow * KSTR + cseg];
    bf16* vd = &Vt[buf][crow * KSTR + cseg];
#pragma unroll
    for (int c = 0; c < 4; ++c) {
      async_copy16(ks + c * 8, kd + c * 8);
      async_copy16(vs + c * 8, vd + c * 8);
    }
  };

  const int tend = causal ? (qbase + 64) : Tlen;
  const int nch = tend >> 6;
  issue_chunk(0, 0);
  for (int ci = 0; ci < nch; ++ci) {
    const int cur = ci & 1;
    const int tc = ci << 6;
    wait_async_all();
    __syncthreads();
    if (ci + 1 < nch) issue_chunk((ci + 1) << 6, cur ^ 1);

    // Scores S = Q * K^T : B col n = key t (K row t contiguous d = B column)
    U8f sf[4];
#pragma unroll
    for (int j = 0; j < 4; ++j) {
#pragma unroll
      for (int r = 0; r < 8; ++r) sf[j].f[r] = 0.f;
      const bf16* krow = &Ks[cur][(j * 16 + l15) * KSTR];
      sf[j].v = wmma_bf16f32(aq0, load_b_frag(krow, lh), sf[j].v);
      sf[j].v = wmma_bf16f32(aq1, load_b_frag(krow + 32, lh), sf[j].v);
    }

    if (causal) {
#pragma unroll
      for (int j = 0; j < 4; ++j)
#pragma unroll
        for (int r = 0; r < 8; ++r) {
          int t = tc + j * 16 + l15;
          int s = qrow0 + r + 8 * lh;
          if (t > s) sf[j].f[r] = -3.0e38f;
        }
    }

    // Online softmax (per-row stats live in the 16-lane half owning that row)
#pragma unroll
    for (int r = 0; r < 8; ++r) {
      float rm = fmaxf(fmaxf(sf[0].f[r], sf[1].f[r]),
                       fmaxf(sf[2].f[r], sf[3].f[r]));
#pragma unroll
      for (int off = 8; off >= 1; off >>= 1) rm = fmaxf(rm, __shfl_xor(rm, off, 16));
      float mn = fmaxf(mrow[r], rm);
      float scale = __expf(mrow[r] - mn);
      mrow[r] = mn;
      float rs = 0.f;
#pragma unroll
      for (int j = 0; j < 4; ++j) {
        float p = __expf(sf[j].f[r] - mn);
        sf[j].f[r] = p;
        rs += p;
      }
#pragma unroll
      for (int off = 8; off >= 1; off >>= 1) rs += __shfl_xor(rs, off, 16);
      lrow[r] = lrow[r] * scale + rs;
#pragma unroll
      for (int dn = 0; dn < 4; ++dn) o[dn].f[r] *= scale;
    }

    // Re-layout P (C-frag layout) -> A-frag layout via per-wave LDS tile
#pragma unroll
    for (int j = 0; j < 4; ++j)
#pragma unroll
      for (int r = 0; r < 8; ++r)
        Pw[(r + 8 * lh) * KSTR + j * 16 + l15] = (bf16)sf[j].f[r];
    asm volatile("s_wait_dscnt 0" ::: "memory");  // wave-local DS store->load

    const bf16* prow = &Pw[l15 * KSTR];
    v16bf ap0 = load_a_frag(prow, lh);       // keys [0,32)
    v16bf ap1 = load_a_frag(prow + 32, lh);  // keys [32,64)
#pragma unroll
    for (int dn = 0; dn < 4; ++dn) {
      const bf16* vcol = &Vt[cur][(dn * 16 + l15) * KSTR];
      o[dn].v = wmma_bf16f32(ap0, load_b_frag(vcol, lh), o[dn].v);
      o[dn].v = wmma_bf16f32(ap1, load_b_frag(vcol + 32, lh), o[dn].v);
    }
  }

  // Finalize: O /= l, write bf16 [B, S, H*D]
#pragma unroll
  for (int r = 0; r < 8; ++r) {
    float inv = 1.f / lrow[r];
#pragma unroll
    for (int dn = 0; dn < 4; ++dn) o[dn].f[r] *= inv;
  }
#pragma unroll
  for (int dn = 0; dn < 4; ++dn)
#pragma unroll
    for (int r = 0; r < 8; ++r) {
      int s = qrow0 + r + 8 * lh;
      Out[((size_t)b * Slen + s) * (Hh * D) + h * D + dn * 16 + l15] =
          (bf16)o[dn].f[r];
    }
}

// ---------------------------------------------------------------------------
// Launcher
// ---------------------------------------------------------------------------
extern "C" void kernel_launch(void* const* d_in, const int* in_sizes, int n_in,
                              void* d_out, int out_size, void* d_ws,
                              size_t ws_size, hipStream_t stream) {
  (void)in_sizes; (void)n_in; (void)out_size; (void)ws_size;
  const int Bb = 4, S = 1024, T = 1024, E = 1024, H = 16, D = 64, F = 4096;
  const int M = Bb * S;

  const float* x     = (const float*)d_in[0];
  const float* enc   = (const float*)d_in[1];
  const float* ln1_g = (const float*)d_in[2];
  const float* ln1_b = (const float*)d_in[3];
  const float* ln2_g = (const float*)d_in[4];
  const float* ln2_b = (const float*)d_in[5];
  const float* ln3_g = (const float*)d_in[6];
  const float* ln3_b = (const float*)d_in[7];
  const float* Wq1 = (const float*)d_in[8];  const float* bq1 = (const float*)d_in[9];
  const float* Wk1 = (const float*)d_in[10]; const float* bk1 = (const float*)d_in[11];
  const float* Wv1 = (const float*)d_in[12]; const float* bv1 = (const float*)d_in[13];
  const float* Wo1 = (const float*)d_in[14]; const float* bo1 = (const float*)d_in[15];
  const float* Wq2 = (const float*)d_in[16]; const float* bq2 = (const float*)d_in[17];
  const float* Wk2 = (const float*)d_in[18]; const float* bk2 = (const float*)d_in[19];
  const float* Wv2 = (const float*)d_in[20]; const float* bv2 = (const float*)d_in[21];
  const float* Wo2 = (const float*)d_in[22]; const float* bo2 = (const float*)d_in[23];
  const float* Wup = (const float*)d_in[24]; const float* bup = (const float*)d_in[25];
  const float* Wdn = (const float*)d_in[26]; const float* bdn = (const float*)d_in[27];

  char* ws = (char*)d_ws;
  size_t off = 0;
  auto alloc = [&](size_t bytes) -> void* {
    void* p = ws + off;
    off = (off + bytes + 255) & ~(size_t)255;
    return p;
  };
  const size_t EHD = (size_t)E * H * D;  // 1M elems
  bf16* wq1b = (bf16*)alloc(EHD * 2);
  bf16* wk1b = (bf16*)alloc(EHD * 2);
  bf16* wv1b = (bf16*)alloc(EHD * 2);
  bf16* wo1b = (bf16*)alloc((size_t)E * E * 2);
  bf16* wq2b = (bf16*)alloc(EHD * 2);
  bf16* wk2b = (bf16*)alloc(EHD * 2);
  bf16* wv2b = (bf16*)alloc(EHD * 2);
  bf16* wo2b = (bf16*)alloc((size_t)E * E * 2);
  bf16* wupb = (bf16*)alloc((size_t)E * F * 2);
  bf16* wdnb = (bf16*)alloc((size_t)F * E * 2);
  bf16* encb = (bf16*)alloc((size_t)Bb * T * E * 2);
  bf16* hlnb = (bf16*)alloc((size_t)M * E * 2);
  bf16* qb   = (bf16*)alloc((size_t)Bb * H * S * D * 2);
  bf16* kb   = (bf16*)alloc((size_t)Bb * H * T * D * 2);
  bf16* vb   = (bf16*)alloc((size_t)Bb * H * T * D * 2);
  bf16* attb = (bf16*)alloc((size_t)M * E * 2);
  float* x1  = (float*)alloc((size_t)M * E * 4);
  float* x2  = (float*)alloc((size_t)M * E * 4);
  bf16* ffb  = (bf16*)alloc((size_t)M * F * 2);

  const int nW = (int)EHD;  // 1M
  dim3 b256(256);
  // Weight conversions (all pre-transposed to Bt[n][k])
  cvt_wqkv_t_kernel<<<nW / 256, b256, 0, stream>>>(Wq1, wq1b);
  cvt_wqkv_t_kernel<<<nW / 256, b256, 0, stream>>>(Wk1, wk1b);
  cvt_wqkv_t_kernel<<<nW / 256, b256, 0, stream>>>(Wv1, wv1b);
  cvt_wqkv_t_kernel<<<nW / 256, b256, 0, stream>>>(Wq2, wq2b);
  cvt_wqkv_t_kernel<<<nW / 256, b256, 0, stream>>>(Wk2, wk2b);
  cvt_wqkv_t_kernel<<<nW / 256, b256, 0, stream>>>(Wv2, wv2b);
  cvt_mat_t_kernel<<<(E * E) / 256, b256, 0, stream>>>(Wo1, wo1b, E, E);
  cvt_mat_t_kernel<<<(E * E) / 256, b256, 0, stream>>>(Wo2, wo2b, E, E);
  cvt_mat_t_kernel<<<(E * F) / 256, b256, 0, stream>>>(Wup, wupb, E, F);
  cvt_mat_t_kernel<<<(F * E) / 256, b256, 0, stream>>>(Wdn, wdnb, F, E);
  cvt_f32_bf16_kernel<<<(Bb * T * E) / 256, b256, 0, stream>>>(enc, encb, Bb * T * E);

  dim3 gN8(1024 / 128, M / 128);   // N=1024 GEMMs
  dim3 gN32(4096 / 128, M / 128);  // N=4096 GEMM
  dim3 gAttn(S / 64, H, Bb);

  // ---- self-attention block ----
  ln_bf16_kernel<<<M, b256, 0, stream>>>(x, ln1_g, ln1_b, hlnb, E);
  gemm_bf16_kernel<MODE_QKV><<<gN8, b256, 0, stream>>>(hlnb, wq1b, bq1, nullptr, qb, M, E, E);
  gemm_bf16_kernel<MODE_QKV><<<gN8, b256, 0, stream>>>(hlnb, wk1b, bk1, nullptr, kb, M, E, E);
  gemm_bf16_kernel<MODE_VT ><<<gN8, b256, 0, stream>>>(hlnb, wv1b, bv1, nullptr, vb, M, E, E);
  attn_kernel<<<gAttn, dim3(128), 0, stream>>>(qb, kb, vb, attb, S, T, 1);
  gemm_bf16_kernel<MODE_RES><<<gN8, b256, 0, stream>>>(attb, wo1b, bo1, x, x1, M, E, E);

  // ---- cross-attention block ----
  ln_bf16_kernel<<<M, b256, 0, stream>>>(x1, ln2_g, ln2_b, hlnb, E);
  gemm_bf16_kernel<MODE_QKV><<<gN8, b256, 0, stream>>>(hlnb, wq2b, bq2, nullptr, qb, M, E, E);
  gemm_bf16_kernel<MODE_QKV><<<gN8, b256, 0, stream>>>(encb, wk2b, bk2, nullptr, kb, M, E, E);
  gemm_bf16_kernel<MODE_VT ><<<gN8, b256, 0, stream>>>(encb, wv2b, bv2, nullptr, vb, M, E, E);
  attn_kernel<<<gAttn, dim3(128), 0, stream>>>(qb, kb, vb, attb, S, T, 0);
  gemm_bf16_kernel<MODE_RES><<<gN8, b256, 0, stream>>>(attb, wo2b, bo2, x1, x2, M, E, E);

  // ---- FFN block ----
  ln_bf16_kernel<<<M, b256, 0, stream>>>(x2, ln3_g, ln3_b, hlnb, E);
  gemm_bf16_kernel<MODE_GELU><<<gN32, b256, 0, stream>>>(hlnb, wupb, bup, nullptr, ffb, M, F, E);
  gemm_bf16_kernel<MODE_RES><<<gN8, b256, 0, stream>>>(ffb, wdnb, bdn, x2, (float*)d_out, M, E, F);
}
